// PrettyRRN_45346264711546
// MI455X (gfx1250) — matrix operand order repaired and checked by reading, not compiled, not run on hardware
//
#include <hip/hip_runtime.h>
#include <hip/hip_bf16.h>

typedef __attribute__((ext_vector_type(16))) _Float16 v16h;
typedef __attribute__((ext_vector_type(8)))  float    v8f;
typedef __attribute__((ext_vector_type(2)))  _Float16 h2;
typedef _Float16 h16;

// ---------------------------------------------------------------------------
// WMMA helper: D = A(16x32 f16) * B(32x16 f16) + C(16x16 f32)
// ---------------------------------------------------------------------------
static __device__ __forceinline__ v8f wmma_f16(v16h a, v16h b, v8f c) {
  return __builtin_amdgcn_wmma_f32_16x16x32_f16(false, a, false, b, (short)0, c,
                                                false, false);
}

// A fragment (16x32 f16) from a row-major f16 tile [16 x stride] at p.
// ISA layout: lane L holds row (L&15); halves 0..7 -> K=kb+0..7, halves 8..15
// -> K=kb+16..23 with kb=(L>>4)*8.
static __device__ __forceinline__ v16h load_a_frag(const h16* p, int stride,
                                                   int kc, int lane) {
  const int r  = lane & 15;
  const int kb = (lane >> 4) * 8;
  const h16* q = p + r * stride + kc * 32 + kb;
  v16h out;
  ((uint4*)&out)[0] = *(const uint4*)(q);        // K = kb+0 .. kb+7
  ((uint4*)&out)[1] = *(const uint4*)(q + 16);   // K = kb+16 .. kb+23
  return out;
}

// B fragment (32x16 f16) from transposed weights wt[N][K] (row-major over n).
// ISA layout: lane L holds column (L&15); halves 0..7 -> K=kb+0..7, halves 8..15
// -> K=kb+16..23 with kb=(L>>4)*8.
static __device__ __forceinline__ v16h load_b_frag(const h16* wt, int K,
                                                   int ntile, int kc, int lane) {
  const int n  = ntile * 16 + (lane & 15);
  const int kb = (lane >> 4) * 8;
  const h16* q = wt + n * K + kc * 32 + kb;
  v16h out;
  ((uint4*)&out)[0] = *(const uint4*)(q);
  ((uint4*)&out)[1] = *(const uint4*)(q + 16);
  return out;
}

// Block-cooperative: wt[n][k] = (f16) W[row0+k][n], zero-padded for k>=Kvalid.
// K = 1<<Kbits.
static __device__ __forceinline__ void fill_wt(h16* wt, const float* W,
                                               int Kbits, int N, int row0,
                                               int Kvalid, int tid, int nthr) {
  const int K = 1 << Kbits;
  for (int idx = tid; idx < N * K; idx += nthr) {
    const int n = idx >> Kbits;
    const int k = idx & (K - 1);
    float v = (k < Kvalid) ? W[(size_t)(row0 + k) * N + n] : 0.f;
    wt[idx] = (h16)v;
  }
}

// Copy one staged [16][128] f16 tile (per-wave LDS) to global, coalesced b128.
static __device__ __forceinline__ void copy_tile_out(const h16* st, h16* g,
                                                     int lane) {
  const uint4* s = (const uint4*)st;
  uint4* d = (uint4*)g;
#pragma unroll
  for (int i = 0; i < 8; ++i) d[lane + i * 32] = s[lane + i * 32];
}

// ---------------------------------------------------------------------------
// Kernel 1: per-node pre-MLP and a = x0@W1a, bv = x0@W1b
// ---------------------------------------------------------------------------
__global__ void __launch_bounds__(256) k_pre(
    const int* __restrict__ anchors, const int* __restrict__ njumps,
    const float* __restrict__ positions, const int* __restrict__ colors,
    const int* __restrict__ markers,
    const float* __restrict__ preW1, const float* __restrict__ preB1,
    const float* __restrict__ preW2, const float* __restrict__ preB2,
    const float* __restrict__ msgW1,
    h16* __restrict__ x0g, h16* __restrict__ ag, h16* __restrict__ bvg,
    int tiles_per_wave) {
  extern __shared__ __attribute__((aligned(16))) char smem1[];
  h16*   preW1t   = (h16*)smem1;                 // [128][64]
  h16*   preW2t   = preW1t + 128 * 64;           // [128][128]
  h16*   w1aT     = preW2t + 128 * 128;          // [128][128]
  h16*   w1bT     = w1aT + 128 * 128;            // [128][128]
  float* b1s      = (float*)(w1bT + 128 * 128);  // [128]
  float* b2s      = b1s + 128;                   // [128]
  h16*   featAll  = (h16*)(b2s + 128);           // 8 x [16][64]
  h16*   stageAll = featAll + 8 * 16 * 64;       // 8 x [16][128]

  const int tid = threadIdx.x, lane = tid & 31, wave = tid >> 5;
  const int nlo = lane & 15, hi = lane >> 4;

  fill_wt(preW1t, preW1, 6, 128, 0, 42, tid, 256);
  fill_wt(preW2t, preW2, 7, 128, 0, 128, tid, 256);
  fill_wt(w1aT, msgW1, 7, 128, 0, 128, tid, 256);
  fill_wt(w1bT, msgW1, 7, 128, 128, 128, tid, 256);
  if (tid < 128) { b1s[tid] = preB1[tid]; b2s[tid] = preB2[tid]; }
  __syncthreads();

  h16* feat = featAll + wave * 16 * 64;
  h16* st   = stageAll + wave * 16 * 128;

  for (int it = 0; it < tiles_per_wave; ++it) {
    const int tile = (blockIdx.x * tiles_per_wave + it) * 8 + wave;
    const int nb = tile * 16;

    // ---- build feature tile [16][64] (42 real dims + zero pad) ----
    {
      const int r = lane >> 1;
      const int node = nb + r, g = node >> 3;
      const int col = colors[node], mrk = markers[node] - 8;
      const int anc = anchors[g], nj = njumps[g];
      const float px = positions[2 * node], py = positions[2 * node + 1];
      const int k0 = (lane & 1) * 32;
#pragma unroll
      for (int t = 0; t < 32; ++t) {
        const int k = k0 + t;
        float val = 0.f;
        if (k == 0)       val = px;
        else if (k == 1)  val = py;
        else if (k < 10)  val = (col == k - 2)  ? 1.f : 0.f;
        else if (k < 18)  val = (mrk == k - 10) ? 1.f : 0.f;
        else if (k < 34)  val = (anc == k - 18) ? 1.f : 0.f;
        else if (k < 42)  val = (nj  == k - 34) ? 1.f : 0.f;
        feat[r * 64 + k] = (h16)val;
      }
    }
    __syncthreads();

    // ---- pre layer 1: st = relu(feat @ preW1 + b1) ----
    {
      v16h a0 = load_a_frag(feat, 64, 0, lane);
      v16h a1 = load_a_frag(feat, 64, 1, lane);
      v8f acc[8];
#pragma unroll
      for (int nt = 0; nt < 8; ++nt) {
        v8f c = {};
        c = wmma_f16(a0, load_b_frag(preW1t, 64, nt, 0, lane), c);
        c = wmma_f16(a1, load_b_frag(preW1t, 64, nt, 1, lane), c);
        acc[nt] = c;
      }
#pragma unroll
      for (int nt = 0; nt < 8; ++nt) {
        const float bb = b1s[nt * 16 + nlo];
#pragma unroll
        for (int v = 0; v < 8; ++v)
          st[(v + 8 * hi) * 128 + nt * 16 + nlo] =
              (h16)fmaxf(acc[nt][v] + bb, 0.f);
      }
    }
    __syncthreads();

    // ---- pre layer 2: x0 = st @ preW2 + b2 ----
    v8f x0acc[8];
    {
      v16h xa[4];
#pragma unroll
      for (int c = 0; c < 4; ++c) xa[c] = load_a_frag(st, 128, c, lane);
#pragma unroll
      for (int nt = 0; nt < 8; ++nt) {
        v8f acc = {};
#pragma unroll
        for (int c = 0; c < 4; ++c)
          acc = wmma_f16(xa[c], load_b_frag(preW2t, 128, nt, c, lane), acc);
        const float bb = b2s[nt * 16 + nlo];
#pragma unroll
        for (int v = 0; v < 8; ++v) acc[v] += bb;
        x0acc[nt] = acc;
      }
    }
    __syncthreads();

    // stage x0 (f16) and store to global coalesced
#pragma unroll
    for (int nt = 0; nt < 8; ++nt) {
#pragma unroll
      for (int v = 0; v < 8; ++v)
        st[(v + 8 * hi) * 128 + nt * 16 + nlo] = (h16)x0acc[nt][v];
    }
    __syncthreads();
    copy_tile_out(st, x0g + (size_t)nb * 128, lane);

    // ---- a = x0 @ W1a, bv = x0 @ W1b (bias b1 applied later, inside relu) ----
    // A-fragments of x0 are lifted to registers first, so the per-wave stage
    // tile can be reused to emit a and bv with coalesced b128 stores.
    {
      v16h xa[4];
#pragma unroll
      for (int c = 0; c < 4; ++c) xa[c] = load_a_frag(st, 128, c, lane);

      // a = x0 @ W1a -> stage -> global
      {
        v8f aa[8];
#pragma unroll
        for (int nt = 0; nt < 8; ++nt) {
          v8f acc = {};
#pragma unroll
          for (int c = 0; c < 4; ++c)
            acc = wmma_f16(xa[c], load_b_frag(w1aT, 128, nt, c, lane), acc);
          aa[nt] = acc;
        }
#pragma unroll
        for (int nt = 0; nt < 8; ++nt) {
#pragma unroll
          for (int v = 0; v < 8; ++v)
            st[(v + 8 * hi) * 128 + nt * 16 + nlo] = (h16)aa[nt][v];
        }
        copy_tile_out(st, ag + (size_t)nb * 128, lane);
      }
      // bv = x0 @ W1b -> stage -> global
      {
        v8f bb[8];
#pragma unroll
        for (int nt = 0; nt < 8; ++nt) {
          v8f acc = {};
#pragma unroll
          for (int c = 0; c < 4; ++c)
            acc = wmma_f16(xa[c], load_b_frag(w1bT, 128, nt, c, lane), acc);
          bb[nt] = acc;
        }
#pragma unroll
        for (int nt = 0; nt < 8; ++nt) {
#pragma unroll
          for (int v = 0; v < 8; ++v)
            st[(v + 8 * hi) * 128 + nt * 16 + nlo] = (h16)bb[nt][v];
        }
        copy_tile_out(st, bvg + (size_t)nb * 128, lane);
      }
    }
    __syncthreads();
  }
}

// ---------------------------------------------------------------------------
// Kernel 2: per 16-node tile (2 graphs): packed-f16 edge relu-sum, upd GEMM,
// post-MLP, per-graph reduction.
// ---------------------------------------------------------------------------
__global__ void __launch_bounds__(256) k_msgpost(
    const float* __restrict__ positions,
    const float* __restrict__ msgW1, const float* __restrict__ msgB1,
    const float* __restrict__ msgW2, const float* __restrict__ msgB2,
    const float* __restrict__ postW1, const float* __restrict__ postB1,
    const float* __restrict__ postW2, const float* __restrict__ postB2,
    const h16* __restrict__ x0g, const h16* __restrict__ ag,
    const h16* __restrict__ bvg, h16* __restrict__ gsum, int tiles_per_wave) {
  extern __shared__ __attribute__((aligned(16))) char smem2[];
  h16*   w2T   = (h16*)smem2;                    // [128][128]
  h16*   pw1aT = w2T + 128 * 128;                // [128][128]
  h16*   pw1bT = pw1aT + 128 * 128;              // [128][128]
  h16*   pw2T  = pw1bT + 128 * 128;              // [128][128]
  h16*   wdh   = pw2T + 128 * 128;               // [128] msg_W1 row 256 (f16)
  h16*   mb1h  = wdh + 128;                      // [128] msg_b1 (f16)
  float* mb2   = (float*)(mb1h + 128);           // [128]
  float* pb1   = mb2 + 128;
  float* pb2   = pb1 + 128;
  h16*   bvAll = (h16*)(pb2 + 128);              // 8 x [16][128]
  h16*   stAll = bvAll + 8 * 16 * 128;           // 8 x [16][128]
  float* dAll  = (float*)(stAll + 8 * 16 * 128); // 8 x [16][8]

  const int tid = threadIdx.x, lane = tid & 31, wave = tid >> 5;
  const int nlo = lane & 15, hi = lane >> 4;
  const int r = nlo, kb = hi * 8;

  fill_wt(w2T, msgW2, 7, 128, 0, 128, tid, 256);
  fill_wt(pw1aT, postW1, 7, 128, 0, 128, tid, 256);
  fill_wt(pw1bT, postW1, 7, 128, 128, 128, tid, 256);
  fill_wt(pw2T, postW2, 7, 128, 0, 128, tid, 256);
  if (tid < 128) {
    wdh[tid]  = (h16)msgW1[256 * 128 + tid];
    mb1h[tid] = (h16)msgB1[tid];
    mb2[tid] = msgB2[tid];
    pb1[tid] = postB1[tid];
    pb2[tid] = postB2[tid];
  }
  __syncthreads();

  h16*   bvT  = bvAll + wave * 16 * 128;
  h16*   st   = stAll + wave * 16 * 128;
  float* dmat = dAll + wave * 128;

  for (int it = 0; it < tiles_per_wave; ++it) {
    const int tile = (blockIdx.x * tiles_per_wave + it) * 8 + wave;
    const int nb = tile * 16;
    const int g0 = tile * 2;

    // ---- stage bv tile into LDS (coalesced) ----
    {
      const uint4* s = (const uint4*)(bvg + (size_t)nb * 128);
      uint4* d = (uint4*)bvT;
#pragma unroll
      for (int i = 0; i < 8; ++i) d[lane + i * 32] = s[lane + i * 32];
    }
    // ---- distance matrix (within-graph) ----
#pragma unroll
    for (int i = 0; i < 4; ++i) {
      const int e = lane * 4 + i;          // 0..127
      const int rr = e >> 3, jj = e & 7;
      const int j = (rr & 8) | jj;
      const float dx = positions[2 * (nb + rr)]     - positions[2 * (nb + j)];
      const float dy = positions[2 * (nb + rr) + 1] - positions[2 * (nb + j) + 1];
      dmat[rr * 8 + jj] = sqrtf(dx * dx + dy * dy);
    }
    __syncthreads();

    // ---- a fragments (lane layout == hsum layout) ----
    v16h aF[4];
#pragma unroll
    for (int c = 0; c < 4; ++c)
      aF[c] = load_a_frag(ag + (size_t)nb * 128, 128, c, lane);

    // neighbors of row r (skip self); f16 splat of each distance
    h2 d2[7]; int jrow[7];
#pragma unroll
    for (int q = 0; q < 7; ++q) {
      const int jj = q + ((q >= (r & 7)) ? 1 : 0);
      jrow[q] = (r & 8) | jj;
      const h16 dh = (h16)dmat[r * 8 + jj];
      d2[q] = (h2){dh, dh};
    }

    // ---- hsum[i] = sum_j relu(a[i] + bv[j] + d_ij*w_d + b1) ----
    // Packed f16 math; channel pairs map 1:1 onto A-fragment half pairs.
    v16h hF[4];
    const h2 zero2 = {};
#pragma unroll
    for (int c = 0; c < 4; ++c) {
#pragma unroll
      for (int tp = 0; tp < 8; ++tp) {
        const int kp = c * 32 + kb + ((tp < 4) ? 2 * tp : 2 * tp + 8);
        const h2 ap  = ((const h2*)&aF[c])[tp];
        const h2 b1p = *(const h2*)(mb1h + kp);
        const h2 wdp = *(const h2*)(wdh + kp);
        const h2 base2 = ap + b1p;
        h2 accp = zero2;
#pragma unroll
        for (int q = 0; q < 7; ++q) {
          const h2 bvp = *(const h2*)(bvT + jrow[q] * 128 + kp);
          h2 v = d2[q] * wdp + base2;   // v_pk_fma_f16
          v = v + bvp;                  // v_pk_add_f16
          accp = accp + __builtin_elementwise_max(v, zero2);  // pk_max + pk_add
        }
        ((h2*)&hF[c])[tp] = accp;
      }
    }

    // ---- upd = hsum @ W2 + 7*b2  -> stage ----
    {
      v8f uacc[8];
#pragma unroll
      for (int nt = 0; nt < 8; ++nt) {
        v8f acc = {};
#pragma unroll
        for (int c = 0; c < 4; ++c)
          acc = wmma_f16(hF[c], load_b_frag(w2T, 128, nt, c, lane), acc);
        uacc[nt] = acc;
      }
#pragma unroll
      for (int nt = 0; nt < 8; ++nt) {
        const float bb = 7.f * mb2[nt * 16 + nlo];
#pragma unroll
        for (int v = 0; v < 8; ++v)
          st[(v + 8 * hi) * 128 + nt * 16 + nlo] = (h16)(uacc[nt][v] + bb);
      }
    }
    __syncthreads();

    // ---- post hidden = relu(upd@pW1a + x0@pW1b + pb1) -> stage ----
    {
      v16h uF[4], xF[4];
#pragma unroll
      for (int c = 0; c < 4; ++c) {
        uF[c] = load_a_frag(st, 128, c, lane);
        xF[c] = load_a_frag(x0g + (size_t)nb * 128, 128, c, lane);
      }
      v8f pacc[8];
#pragma unroll
      for (int nt = 0; nt < 8; ++nt) {
        v8f acc = {};
#pragma unroll
        for (int c = 0; c < 4; ++c)
          acc = wmma_f16(uF[c], load_b_frag(pw1aT, 128, nt, c, lane), acc);
#pragma unroll
        for (int c = 0; c < 4; ++c)
          acc = wmma_f16(xF[c], load_b_frag(pw1bT, 128, nt, c, lane), acc);
        pacc[nt] = acc;
      }
#pragma unroll
      for (int nt = 0; nt < 8; ++nt) {
        const float bb = pb1[nt * 16 + nlo];
#pragma unroll
        for (int v = 0; v < 8; ++v)
          st[(v + 8 * hi) * 128 + nt * 16 + nlo] =
              (h16)fmaxf(pacc[nt][v] + bb, 0.f);
      }
    }
    __syncthreads();

    // ---- y = hidden @ pW2 + pb2, reduce 8 rows per graph -> gsum (f16) ----
    {
      v16h hF2[4];
#pragma unroll
      for (int c = 0; c < 4; ++c) hF2[c] = load_a_frag(st, 128, c, lane);
      v8f yacc[8];
#pragma unroll
      for (int nt = 0; nt < 8; ++nt) {
        v8f acc = {};
#pragma unroll
        for (int c = 0; c < 4; ++c)
          acc = wmma_f16(hF2[c], load_b_frag(pw2T, 128, nt, c, lane), acc);
        yacc[nt] = acc;
      }
#pragma unroll
      for (int nt = 0; nt < 8; ++nt) {
        float s = 8.f * pb2[nt * 16 + nlo];
#pragma unroll
        for (int v = 0; v < 8; ++v) s += yacc[nt][v];
        gsum[(size_t)(g0 + hi) * 128 + nt * 16 + nlo] = (h16)s;
      }
    }
    __syncthreads();
  }
}

// ---------------------------------------------------------------------------
// Kernel 3: out-MLP on per-graph features: [bs,128] -> relu 128 -> 16
// ---------------------------------------------------------------------------
__global__ void __launch_bounds__(256) k_out(
    const float* __restrict__ outW1, const float* __restrict__ outB1,
    const float* __restrict__ outW2, const float* __restrict__ outB2,
    const h16* __restrict__ gsum, float* __restrict__ out, int tiles_per_wave) {
  extern __shared__ __attribute__((aligned(16))) char smem3[];
  h16*   ow1T  = (h16*)smem3;                 // [128][128]
  h16*   ow2T  = ow1T + 128 * 128;            // [16][128]
  float* ob1   = (float*)(ow2T + 16 * 128);   // [128]
  float* ob2   = ob1 + 128;                   // [16]
  h16*   stAll = (h16*)(ob2 + 16);            // 8 x [16][128]

  const int tid = threadIdx.x, lane = tid & 31, wave = tid >> 5;
  const int nlo = lane & 15, hi = lane >> 4;

  fill_wt(ow1T, outW1, 7, 128, 0, 128, tid, 256);
  fill_wt(ow2T, outW2, 7, 16, 0, 128, tid, 256);
  if (tid < 128) ob1[tid] = outB1[tid];
  if (tid < 16)  ob2[tid] = outB2[tid];
  __syncthreads();

  h16* st = stAll + wave * 16 * 128;

  for (int it = 0; it < tiles_per_wave; ++it) {
    const int tile = (blockIdx.x * tiles_per_wave + it) * 8 + wave;
    const int gb = tile * 16;

    v16h gF[4];
#pragma unroll
    for (int c = 0; c < 4; ++c)
      gF[c] = load_a_frag(gsum + (size_t)gb * 128, 128, c, lane);
    {
      v8f acc[8];
#pragma unroll
      for (int nt = 0; nt < 8; ++nt) {
        v8f c8 = {};
#pragma unroll
        for (int c = 0; c < 4; ++c)
          c8 = wmma_f16(gF[c], load_b_frag(ow1T, 128, nt, c, lane), c8);
        acc[nt] = c8;
      }
#pragma unroll
      for (int nt = 0; nt < 8; ++nt) {
        const float bb = ob1[nt * 16 + nlo];
#pragma unroll
        for (int v = 0; v < 8; ++v)
          st[(v + 8 * hi) * 128 + nt * 16 + nlo] =
              (h16)fmaxf(acc[nt][v] + bb, 0.f);
      }
    }
    __syncthreads();

    {
      v16h hF[4];
#pragma unroll
      for (int c = 0; c < 4; ++c) hF[c] = load_a_frag(st, 128, c, lane);
      v8f acc = {};
#pragma unroll
      for (int c = 0; c < 4; ++c)
        acc = wmma_f16(hF[c], load_b_frag(ow2T, 128, 0, c, lane), acc);
      const float bb = ob2[nlo];
#pragma unroll
      for (int v = 0; v < 8; ++v)
        out[(size_t)(gb + v + 8 * hi) * 16 + nlo] = acc[v] + bb;
    }
    __syncthreads();
  }
}

// ---------------------------------------------------------------------------
extern "C" void kernel_launch(void* const* d_in, const int* in_sizes, int n_in,
                              void* d_out, int out_size, void* d_ws,
                              size_t ws_size, hipStream_t stream) {
  const int*   anchors   = (const int*)d_in[0];
  const int*   njumps    = (const int*)d_in[1];
  const float* positions = (const float*)d_in[2];
  const int*   colors    = (const int*)d_in[3];
  const int*   markers   = (const int*)d_in[4];
  const float* preW1  = (const float*)d_in[5];
  const float* preB1  = (const float*)d_in[6];
  const float* preW2  = (const float*)d_in[7];
  const float* preB2  = (const float*)d_in[8];
  const float* msgW1  = (const float*)d_in[9];
  const float* msgB1  = (const float*)d_in[10];
  const float* msgW2  = (const float*)d_in[11];
  const float* msgB2  = (const float*)d_in[12];
  const float* postW1 = (const float*)d_in[13];
  const float* postB1 = (const float*)d_in[14];
  const float* postW2 = (const float*)d_in[15];
  const float* postB2 = (const float*)d_in[16];
  const float* outW1  = (const float*)d_in[17];
  const float* outB1  = (const float*)d_in[18];
  const float* outW2  = (const float*)d_in[19];
  const float* outB2  = (const float*)d_in[20];
  float* out = (float*)d_out;

  const int bs = in_sizes[0];            // 16384
  const size_t nnodes = (size_t)bs * 8;  // 131072
  h16* x0g = (h16*)d_ws;
  h16* ag  = x0g + nnodes * 128;
  h16* bvg = ag + nnodes * 128;
  h16* gsm = bvg + nnodes * 128;

  const int ntile16 = bs / 2;  // node tiles of 16 rows (8192)

  {  // kernel 1
    const int tpw = 2;
    const int blocks = ntile16 / (8 * tpw);  // 512
    const size_t sm = (size_t)(128 * 64 + 3 * 128 * 128) * 2 + 2 * 128 * 4 +
                      (size_t)8 * (16 * 64 + 16 * 128) * 2;
    k_pre<<<blocks, 256, sm, stream>>>(anchors, njumps, positions, colors,
                                       markers, preW1, preB1, preW2, preB2,
                                       msgW1, x0g, ag, bvg, tpw);
  }
  {  // kernel 2
    const int tpw = 2;
    const int blocks = ntile16 / (8 * tpw);  // 512
    const size_t sm = (size_t)4 * 128 * 128 * 2 + 2 * 128 * 2 + 3 * 128 * 4 +
                      (size_t)2 * 8 * 16 * 128 * 2 + (size_t)8 * 128 * 4;
    k_msgpost<<<blocks, 256, sm, stream>>>(positions, msgW1, msgB1, msgW2,
                                           msgB2, postW1, postB1, postW2,
                                           postB2, x0g, ag, bvg, gsm, tpw);
  }
  {  // kernel 3
    const int tpw = 1;
    const int gtiles = bs / 16;              // 1024
    const int blocks = gtiles / (8 * tpw);   // 128
    const size_t sm = (size_t)128 * 128 * 2 + (size_t)16 * 128 * 2 + 128 * 4 +
                      16 * 4 + (size_t)8 * 16 * 128 * 2;
    k_out<<<blocks, 256, sm, stream>>>(outW1, outB1, outW2, outB2, gsm, out,
                                       tpw);
  }
}
